// Generator_24008867185217
// MI455X (gfx1250) — compile-verified, matrix operand order checked by reading, not run on hardware
//
#include <hip/hip_runtime.h>
#include <hip/hip_bf16.h>

// ---- problem constants (match reference) ----
#define NB      8
#define NN      64
#define HH      128
#define NHEADS  4
#define HDIM    128
#define INNER   512
#define EE      8
#define NNOISE  128
#define NL      8
#define NEDGE   4032
#define HALFE   2016
#define ATOMN   9
#define OTHERN  15
#define BLK     256
#define PAN_BYTES (HDIM * HDIM * 4)  // 64 KB per staged weight panel

typedef __attribute__((ext_vector_type(16))) _Float16     v16h;
typedef __attribute__((ext_vector_type(8)))  float        v8f;
typedef __attribute__((ext_vector_type(4)))  unsigned int u32x4;
typedef __attribute__((ext_vector_type(8)))  int          i32x8;
typedef __attribute__((ext_vector_type(4)))  int          i32x4;

#define WMMA16(a, b, c) \
  __builtin_amdgcn_wmma_f32_16x16x32_f16(false, (a), false, (b), (short)0, (c), false, false)

// K-index mapping for 16-bit A/B fragments (ISA 7.12.2, 16-bit A 16x32):
// lane<16 holds K = {0..7, 16..23}; lane>=16 holds K = {8..15, 24..31}
__device__ __forceinline__ int kmap(int j, int lane) {
  int base = (lane & 16) ? 8 : 0;
  return (j < 8) ? (base + j) : (8 + base + j);  // j>=8 -> 16 + base + (j-8)
}

// A fragment: row-major f16 in LDS, M = row0 + (lane&15)
__device__ __forceinline__ v16h frag_a_lds(const _Float16* A, int row0, int k0, int lda, int lane) {
  const _Float16* p = A + (row0 + (lane & 15)) * lda + k0;
  v16h a;
#pragma unroll
  for (int j = 0; j < 16; ++j) a[j] = p[kmap(j, lane)];
  return a;
}

// B fragment (KxN) from f32 LDS panel (row-major [K x ld]), converted to f16
__device__ __forceinline__ v16h frag_b_lds_f32(const float* Bm, int k0, int col, int ld, int lane) {
  const float* p = Bm + k0 * ld + col + (lane & 15);
  v16h b;
#pragma unroll
  for (int j = 0; j < 16; ++j) b[j] = (_Float16)p[kmap(j, lane) * ld];
  return b;
}

// B fragment (KxN) from f16 LDS row-major [K x ld]
__device__ __forceinline__ v16h frag_b_lds(const _Float16* Bm, int k0, int col, int ld, int lane) {
  const _Float16* p = Bm + k0 * ld + col + (lane & 15);
  v16h b;
#pragma unroll
  for (int j = 0; j < 16; ++j) b[j] = p[kmap(j, lane) * ld];
  return b;
}

// B fragment where B[k][n] = M[n][k]  (for Q @ K^T)
__device__ __forceinline__ v16h frag_b_lds_T(const _Float16* Bm, int k0, int col, int ld, int lane) {
  const _Float16* p = Bm + (col + (lane & 15)) * ld + k0;
  v16h b;
#pragma unroll
  for (int j = 0; j < 16; ++j) b[j] = p[kmap(j, lane)];
  return b;
}

__device__ __forceinline__ float sigm(float x) { return 1.f / (1.f + __expf(-x)); }

// ---- Tensor Data Mover: DMA a 128x128 f32 tile (row pitch = stride elems)
// from global memory into LDS at byte offset lds_off. D# built per ISA 8.3/8.4.
__device__ __forceinline__ void tdm_load_tile_f32(const float* src, unsigned lds_off, int stride) {
  unsigned long long ga = (unsigned long long)(size_t)src;
  u32x4 g0 = {0u, 0u, 0u, 0u};
  g0[0] = 1u;                                             // count=1 (user mode)
  g0[1] = lds_off;                                        // lds_addr (bytes)
  g0[2] = (unsigned)ga;                                   // global_addr[31:0]
  g0[3] = (unsigned)((ga >> 32) & 0x1FFFFFFu) | (2u << 30);  // addr[56:32] | type=2
  i32x8 g1 = {0, 0, 0, 0, 0, 0, 0, 0};
  g1[0] = 2 << 16;            // data_size = 4 bytes
  g1[1] = HDIM << 16;         // tensor_dim0[15:0] = 128
  g1[2] = HDIM << 16;         // tensor_dim1[15:0] = 128
  g1[3] = HDIM << 16;         // tile_dim0 = 128
  g1[4] = HDIM;               // tile_dim1 = 128 (tile_dim2 = 0)
  g1[5] = stride;             // tensor_dim0_stride[31:0]
  i32x4 z4 = {0, 0, 0, 0};
#if defined(__clang_major__) && __clang_major__ >= 23
  i32x8 z8 = {0, 0, 0, 0, 0, 0, 0, 0};
  __builtin_amdgcn_tensor_load_to_lds(g0, g1, z4, z4, z8, 0);
#else
  __builtin_amdgcn_tensor_load_to_lds(g0, g1, z4, z4, 0);
#endif
}

// proj = xf16 @ panel + bias  (panel is f32 LDS [128][128]); result f16 -> dst (LDS)
__device__ __forceinline__ void proj_gemm(const _Float16* A, const float* pan,
                                          const float* bias, _Float16* dst,
                                          int wave, int lane) {
  for (int t = wave; t < 32; t += 8) {
    const int mt = (t >> 3) * 16, nt = (t & 7) * 16;
    v8f c = {};
#pragma unroll
    for (int k0 = 0; k0 < HH; k0 += 32)
      c = WMMA16(frag_a_lds(A, mt, k0, HH, lane),
                 frag_b_lds_f32(pan, k0, nt, HDIM, lane), c);
    const int cc = nt + (lane & 15), ro = mt + ((lane & 16) ? 8 : 0);
#pragma unroll
    for (int r = 0; r < 8; ++r)
      dst[(ro + r) * HDIM + cc] = (_Float16)(c[r] + bias[cc]);
  }
}

__global__ __launch_bounds__(BLK) void molgen_kernel(
    const float* __restrict__ noise, const float* __restrict__ edge_attr,
    const int*   __restrict__ edge_index,
    const float* __restrict__ fc1_w, const float* __restrict__ fc1_b,
    const float* __restrict__ q_w, const float* __restrict__ q_b,
    const float* __restrict__ k_w, const float* __restrict__ k_b,
    const float* __restrict__ v_w, const float* __restrict__ v_b,
    const float* __restrict__ e_w,
    const float* __restrict__ skip_w, const float* __restrict__ skip_b,
    const float* __restrict__ atom_w, const float* __restrict__ atom_b,
    const float* __restrict__ other_w, const float* __restrict__ other_b,
    const float* __restrict__ edge_fc_w, const float* __restrict__ edge_fc_b,
    float* __restrict__ out) {
  extern __shared__ char smem[];
  // Double-buffered TDM panels at LDS offsets 0 and 64KB (descriptors use absolute LDS addr).
  float*    pan0 = (float*)smem;            // [128][128] weight panel A (TDM dest)
  float*    pan1 = pan0 + HDIM * HDIM;      // [128][128] weight panel B (TDM dest)
  float*    xf32 = pan1 + HDIM * HDIM;      // [64][128]
  float*    acc  = xf32 + NN * HH;          // [64][128] head-mean accumulator
  float*    sS   = acc + NN * HH;           // [64][64]  scores / exp scratch
  float*    tmat = sS + NN * NN;            // [64][8]   t[d] = e_w_h^T q[d]
  float*    wmat = tmat + NN * EE;          // [64][8]   alpha-weighted edge_attr sums
  _Float16* xf16 = (_Float16*)(wmat + NN * EE);  // [64][128]
  _Float16* qh   = xf16 + NN * HH;          // [64][128] per-head Q
  _Float16* kh   = qh + NN * HH;            // per-head K
  _Float16* vh   = kh + NN * HH;            // per-head V
  _Float16* alp  = vh + NN * HH;            // [64][64] alpha (f16)

  const int b = blockIdx.x, tid = threadIdx.x, lane = tid & 31, wave = tid >> 5;
  const float* ea = edge_attr + (size_t)b * NEDGE * EE;
  const float scale = 0.0883883476483184f;  // 1/sqrt(128)

  // ---- x init: h = relu(noise @ fc1_w + b), broadcast to all 64 nodes ----
  if (tid < HH) {
    float s = fc1_b[tid];
    const float* nz = noise + (size_t)b * NNOISE;
    for (int i = 0; i < NNOISE; ++i) s += nz[i] * fc1_w[i * HH + tid];
    xf32[tid] = s > 0.f ? s : 0.f;
  }
  __syncthreads();
  for (int i = tid + HH; i < NN * HH; i += BLK) xf32[i] = xf32[i & (HH - 1)];
  __syncthreads();

  // ---- 8 TransformerConv layers ----
  for (int l = 0; l < NL; ++l) {
    const float* Wq = q_w + (size_t)l * HH * INNER;
    const float* Wk = k_w + (size_t)l * HH * INNER;
    const float* Wv = v_w + (size_t)l * HH * INNER;
    const float* We = e_w + (size_t)l * EE * INNER;
    const float* Ws = skip_w + (size_t)l * HH * HH;
    const float* bq = q_b + l * INNER;
    const float* bk = k_b + l * INNER;
    const float* bv = v_b + l * INNER;
    const float* bs = skip_b + l * HH;

    for (int i = tid; i < NN * HH; i += BLK) {
      xf16[i] = (_Float16)xf32[i];
      acc[i] = 0.f;
    }
    __builtin_prefetch(We + (size_t)tid * 16, 0, 1);  // global_prefetch path
    __syncthreads();

    for (int hh = 0; hh < NHEADS; ++hh) {
      const int cb = hh * HDIM;

      // --- Q/K/V projections with double-buffered TDM: DMA next panel while
      //     WMMA consumes the current one. Wave 0 issues + waits TENSORcnt;
      //     workgroup barriers publish panels. ---
      if (wave == 0) {
        tdm_load_tile_f32(Wq + cb, 0u, INNER);            // P0 <- Wq panel
        __builtin_amdgcn_s_wait_tensorcnt(0);
      }
      __syncthreads();
      if (wave == 0) tdm_load_tile_f32(Wk + cb, PAN_BYTES, INNER);  // P1 <- Wk (overlaps Q GEMM)
      proj_gemm(xf16, pan0, bq + cb, qh, wave, lane);
      __syncthreads();  // all waves done with P0
      if (wave == 0) {
        __builtin_amdgcn_s_wait_tensorcnt(0);             // P1 landed
        tdm_load_tile_f32(Wv + cb, 0u, INNER);            // P0 <- Wv (overlaps K GEMM)
      }
      __syncthreads();
      proj_gemm(xf16, pan1, bk + cb, kh, wave, lane);
      __syncthreads();  // all waves done with P1
      if (wave == 0) __builtin_amdgcn_s_wait_tensorcnt(0);  // P0 landed
      __syncthreads();
      proj_gemm(xf16, pan0, bv + cb, vh, wave, lane);
      // qh/kh already barrier-published; vh is only read after two more barriers.

      // --- raw S = Q K^T  (dst-major [d][s]) ---
      for (int t = wave; t < 16; t += 8) {
        const int mt = (t >> 2) * 16, nt = (t & 3) * 16;
        v8f c = {};
#pragma unroll
        for (int k0 = 0; k0 < HDIM; k0 += 32)
          c = WMMA16(frag_a_lds(qh, mt, k0, HDIM, lane),
                     frag_b_lds_T(kh, k0, nt, HDIM, lane), c);
        const int cc = nt + (lane & 15), ro = mt + ((lane & 16) ? 8 : 0);
#pragma unroll
        for (int r = 0; r < 8; ++r) sS[(ro + r) * NN + cc] = c[r];
      }
      // --- t[d][e'] = sum_c q[d][c] * e_w[e'][cb+c]  (factored edge term) ---
      for (int i = tid; i < NN * EE; i += BLK) {
        const int d = i >> 3, ep = i & 7;
        const float* wr = We + ep * INNER + cb;
        const _Float16* qr = qh + d * HDIM;
        float s = 0.f;
        for (int c2 = 0; c2 < HDIM; ++c2) s += (float)qr[c2] * wr[c2];
        tmat[i] = s;
      }
      __syncthreads();

      // --- per-dst softmax over incoming edges (s != d) + edge-weight sums ---
      if (tid < NN) {
        const int d = tid;
        float* srow = sS + d * NN;
        const float* trow = tmat + d * EE;
        float mx = -1e30f;
        for (int s2 = 0; s2 < NN; ++s2) {
          if (s2 == d) continue;
          const int eidx = (s2 < d)
              ? (s2 * (2 * NN - s2 - 1) / 2 + (d - s2 - 1))
              : (HALFE + d * (2 * NN - d - 1) / 2 + (s2 - d - 1));
          const float* er = ea + (size_t)eidx * EE;
          float qe = 0.f;
#pragma unroll
          for (int ep = 0; ep < EE; ++ep) qe += er[ep] * trow[ep];
          const float sc = scale * (srow[s2] + qe);
          srow[s2] = sc;
          mx = sc > mx ? sc : mx;
        }
        float sum = 0.f;
        for (int s2 = 0; s2 < NN; ++s2) {
          if (s2 == d) continue;
          const float e2 = __expf(srow[s2] - mx);
          srow[s2] = e2;
          sum += e2;
        }
        const float inv = 1.f / sum;
        float wl[EE];
#pragma unroll
        for (int ep = 0; ep < EE; ++ep) wl[ep] = 0.f;
        for (int s2 = 0; s2 < NN; ++s2) {
          if (s2 == d) { alp[d * NN + s2] = (_Float16)0.f; continue; }
          const float al = srow[s2] * inv;
          alp[d * NN + s2] = (_Float16)al;
          const int eidx = (s2 < d)
              ? (s2 * (2 * NN - s2 - 1) / 2 + (d - s2 - 1))
              : (HALFE + d * (2 * NN - d - 1) / 2 + (s2 - d - 1));
          const float* er = ea + (size_t)eidx * EE;
#pragma unroll
          for (int ep = 0; ep < EE; ++ep) wl[ep] += al * er[ep];
        }
#pragma unroll
        for (int ep = 0; ep < EE; ++ep) wmat[d * EE + ep] = wl[ep];
      }
      __syncthreads();

      // During the last head's aggregation, stream in the skip weights (P1 is idle).
      if (hh == NHEADS - 1 && wave == 0) tdm_load_tile_f32(Ws, PAN_BYTES, HH);

      // --- agg = alpha @ V  (WMMA)  +  (alpha-weighted ea) @ e_w ---
      for (int t = wave; t < 32; t += 8) {
        const int mt = (t >> 3) * 16, nt = (t & 7) * 16;
        v8f c = {};
#pragma unroll
        for (int k0 = 0; k0 < NN; k0 += 32)
          c = WMMA16(frag_a_lds(alp, mt, k0, NN, lane),
                     frag_b_lds(vh, k0, nt, HDIM, lane), c);
        const int cc = nt + (lane & 15), ro = mt + ((lane & 16) ? 8 : 0);
#pragma unroll
        for (int r = 0; r < 8; ++r) {
          const int row = ro + r;
          float eggv = 0.f;
          const float* wr = wmat + row * EE;
#pragma unroll
          for (int ep = 0; ep < EE; ++ep) eggv += wr[ep] * We[ep * INNER + cb + cc];
          acc[row * HDIM + cc] += 0.25f * (c[r] + eggv);  // head mean
        }
      }
      __syncthreads();
    }  // heads

    // --- out = head_mean + x @ skip_w + skip_b, relu -> new x (skip panel in P1) ---
    if (wave == 0) __builtin_amdgcn_s_wait_tensorcnt(0);
    __syncthreads();
    for (int t = wave; t < 32; t += 8) {
      const int mt = (t >> 3) * 16, nt = (t & 7) * 16;
      v8f c = {};
#pragma unroll
      for (int k0 = 0; k0 < HH; k0 += 32)
        c = WMMA16(frag_a_lds(xf16, mt, k0, HH, lane),
                   frag_b_lds_f32(pan1, k0, nt, HDIM, lane), c);
      const int cc = nt + (lane & 15), ro = mt + ((lane & 16) ? 8 : 0);
#pragma unroll
      for (int r = 0; r < 8; ++r) {
        const int row = ro + r;
        const float v2 = c[r] + acc[row * HH + cc] + bs[cc];
        xf32[row * HH + cc] = v2 > 0.f ? v2 : 0.f;
      }
    }
    __syncthreads();
  }  // layers

  // ---- node features: sigmoid(concat(softmax(x@Wa+ba), sigmoid(x@Wo+bo))) ----
  if (tid < NN) {
    const int d = tid;
    const float* xr = xf32 + d * HH;
    float* on = out + ((size_t)b * NN + d) * (ATOMN + OTHERN);
    float lg[ATOMN];
    float mx = -1e30f;
#pragma unroll
    for (int i = 0; i < ATOMN; ++i) {
      float s = atom_b[i];
      for (int c2 = 0; c2 < HH; ++c2) s += xr[c2] * atom_w[c2 * ATOMN + i];
      lg[i] = s;
      mx = s > mx ? s : mx;
    }
    float sum = 0.f;
#pragma unroll
    for (int i = 0; i < ATOMN; ++i) { lg[i] = __expf(lg[i] - mx); sum += lg[i]; }
    const float inv = 1.f / sum;
#pragma unroll
    for (int i = 0; i < ATOMN; ++i) on[i] = sigm(lg[i] * inv);
#pragma unroll
    for (int j = 0; j < OTHERN; ++j) {
      float s = other_b[j];
      for (int c2 = 0; c2 < HH; ++c2) s += xr[c2] * other_w[c2 * OTHERN + j];
      on[ATOMN + j] = sigm(sigm(s));
    }
  }

  // ---- edge features: sigmoid([x[src], x[dst]] @ W_ef + b_ef) ----
  float* oe = out + (size_t)NB * NN * (ATOMN + OTHERN) + (size_t)b * NEDGE * EE;
  for (int e2 = tid; e2 < NEDGE; e2 += BLK) {
    const int s2 = edge_index[e2], d2 = edge_index[NEDGE + e2];
    const float* xs = xf32 + s2 * HH;
    const float* xd = xf32 + d2 * HH;
    float a8[EE];
#pragma unroll
    for (int ep = 0; ep < EE; ++ep) a8[ep] = edge_fc_b[ep];
    for (int c2 = 0; c2 < HH; ++c2) {
      const float vs = xs[c2], vd = xd[c2];
      const float* w1 = edge_fc_w + c2 * EE;
      const float* w2 = edge_fc_w + (HH + c2) * EE;
#pragma unroll
      for (int ep = 0; ep < EE; ++ep) a8[ep] += vs * w1[ep] + vd * w2[ep];
    }
#pragma unroll
    for (int ep = 0; ep < EE; ++ep) oe[(size_t)e2 * EE + ep] = sigm(a8[ep]);
  }
}

extern "C" void kernel_launch(void* const* d_in, const int* in_sizes, int n_in,
                              void* d_out, int out_size, void* d_ws, size_t ws_size,
                              hipStream_t stream) {
  (void)in_sizes; (void)n_in; (void)out_size; (void)d_ws; (void)ws_size;
  const float* noise     = (const float*)d_in[0];
  const float* edge_attr = (const float*)d_in[1];
  const int*   edge_idx  = (const int*)d_in[2];
  const float* fc1_w     = (const float*)d_in[3];
  const float* fc1_b     = (const float*)d_in[4];
  const float* q_w       = (const float*)d_in[5];
  const float* q_b       = (const float*)d_in[6];
  const float* k_w       = (const float*)d_in[7];
  const float* k_b       = (const float*)d_in[8];
  const float* v_w       = (const float*)d_in[9];
  const float* v_b       = (const float*)d_in[10];
  const float* e_w       = (const float*)d_in[11];
  const float* skip_w    = (const float*)d_in[12];
  const float* skip_b    = (const float*)d_in[13];
  const float* atom_w    = (const float*)d_in[14];
  const float* atom_b    = (const float*)d_in[15];
  const float* other_w   = (const float*)d_in[16];
  const float* other_b   = (const float*)d_in[17];
  const float* edge_fc_w = (const float*)d_in[18];
  const float* edge_fc_b = (const float*)d_in[19];

  // LDS: 2x64KB TDM panels + floats (x, acc, S, t, w) + halfs (xf16, Q, K, V, alpha)
  const size_t shmem =
      (size_t)(2 * HDIM * HDIM) * sizeof(float) +
      (size_t)(NN * HH * 2 + NN * NN + NN * EE * 2) * sizeof(float) +
      (size_t)(NN * HH * 4 + NN * NN) * sizeof(_Float16);  // 290816 B < 320 KB/WGP

  molgen_kernel<<<dim3(NB), dim3(BLK), shmem, stream>>>(
      noise, edge_attr, edge_idx, fc1_w, fc1_b, q_w, q_b, k_w, k_b, v_w, v_b,
      e_w, skip_w, skip_b, atom_w, atom_b, other_w, other_b, edge_fc_w,
      edge_fc_b, (float*)d_out);
}
